// BA_Encoder_8297876816439
// MI455X (gfx1250) — matrix operand order repaired and checked by reading, not compiled, not run on hardware
//
#include <hip/hip_runtime.h>
#include <hip/hip_bf16.h>
#include <math.h>

typedef __bf16 bf16_t;
typedef __attribute__((ext_vector_type(8)))  __bf16 v8bf;
typedef __attribute__((ext_vector_type(16))) __bf16 v16bf;
typedef __attribute__((ext_vector_type(8)))  float  v8f;

static inline int cdiv(long a, int b) { return (int)((a + b - 1) / b); }

// ---------------- fills / converts ----------------
__global__ void fill_f32(float* p, float v, int n) {
  int i = blockIdx.x * blockDim.x + threadIdx.x;
  if (i < n) p[i] = v;
}
__global__ void fill_u32(unsigned* p, unsigned v, int n) {
  int i = blockIdx.x * blockDim.x + threadIdx.x;
  if (i < n) p[i] = v;
}
__global__ void cvt_bf16(const float* __restrict__ x, bf16_t* __restrict__ o, long n) {
  long i = blockIdx.x * (long)blockDim.x + threadIdx.x;
  if (i < n) o[i] = (bf16_t)x[i];
}

// Pack W[K,C] (f32, row-major) into per-(tileN, k-group, lane) contiguous bf16 runs
// in the contiguous-k B order (k = 16*half + j) that gemm_wmma also uses for A: elem[(((tileN*kt + kk)*32)+lane)*16 + j]
//   = W[(kk*32 + (lane>>4)*16 + j)*C + tileN*16 + (lane&15)]
__global__ void pack_w(const float* __restrict__ W, bf16_t* __restrict__ out, int K, int C) {
  long i = blockIdx.x * (long)blockDim.x + threadIdx.x;
  if (i >= (long)K * C) return;
  int kt = K >> 5;
  int j = (int)(i & 15);
  long t = i >> 4;
  int lane = (int)(t & 31);
  t >>= 5;
  int kk    = (int)(t % kt);
  int tileN = (int)(t / kt);
  int k   = (kk << 5) + ((lane >> 4) << 4) + j;
  int col = (tileN << 4) + (lane & 15);
  out[i] = (bf16_t)W[(long)k * C + col];
}

// ---------------- WMMA GEMM: H[N,C] = XB[N,K] @ W[K,C], bf16 in / f32 acc ----
// XB: bf16 row-major. WP: pre-packed B layout (pack_w). One wave32 = 16x16 tile.
// Steady state: 4x global_load_b128 + 1x v_wmma_f32_16x16x32_bf16 per K-step.
__global__ void gemm_wmma(const bf16_t* __restrict__ XB, const bf16_t* __restrict__ WP,
                          float* __restrict__ H, int N, int K, int C) {
  int gw   = (int)((blockIdx.x * (long)blockDim.x + threadIdx.x) >> 5);
  int lane = threadIdx.x & 31;
  int tilesC = C >> 4;
  int tileM = gw / tilesC;
  int tileN = gw - tileM * tilesC;
  if (tileM * 16 >= N) return;            // wave-uniform exit (EXEC all-1s at WMMA)
  int half = lane >> 4;
  int row  = tileM * 16 + (lane & 15);
  int kt   = K >> 5;

  const v8bf* xv = (const v8bf*)XB;
  const v8bf* wv = (const v8bf*)WP;
  long abase = ((long)row * K) >> 3;                    // v8bf units
  long bbase = ((long)tileN * kt * 32 + lane) * 2;      // v8bf units

  v8f cacc = {};
  for (int kk = 0; kk < kt; ++kk) {
    if (kk + 1 < kt) __builtin_prefetch(xv + abase + (kk + 1) * 4, 0, 1);
    v8bf a0 = xv[abase + kk * 4 + 2 * half];    // K: k0+16h  .. +8
    v8bf a1 = xv[abase + kk * 4 + 2 * half + 1];// K: k0+16h+8.. +8
    v8bf b0 = wv[bbase + (long)kk * 64];        // K: k0+16h  .. +8
    v8bf b1 = wv[bbase + (long)kk * 64 + 1];    // K: k0+16h+8.. +8
    v16bf a = __builtin_shufflevector(a0, a1, 0,1,2,3,4,5,6,7,8,9,10,11,12,13,14,15);
    v16bf b = __builtin_shufflevector(b0, b1, 0,1,2,3,4,5,6,7,8,9,10,11,12,13,14,15);
    cacc = __builtin_amdgcn_wmma_f32_16x16x32_bf16(false, a, false, b,
                                                   (short)0, cacc, false, false);
  }
#pragma unroll
  for (int r = 0; r < 8; ++r) {
    int m = tileM * 16 + r + (half << 3);       // C/D layout: M = r + 8*half
    H[(long)m * C + (tileN * 16 + (lane & 15))] = cacc[r];
  }
}

// ---------------- wave-dot: out[r] = dot(X[r, 0:K], v) ----------------
__global__ void dot_rows(const float* __restrict__ X, const float* __restrict__ vec,
                         float* __restrict__ out, int N, int K) {
  int r    = (int)((blockIdx.x * (long)blockDim.x + threadIdx.x) >> 5);
  int lane = threadIdx.x & 31;
  if (r >= N) return;
  float s = 0.f;
  for (int k = lane; k < K; k += 32) s += X[(long)r * K + k] * vec[k];
#pragma unroll
  for (int o = 16; o; o >>= 1) s += __shfl_xor(s, o, 32);
  if (lane == 0) out[r] = s;
}

// ------------- ordered-uint encoding for float atomic max -------------
__device__ __forceinline__ unsigned f2ord(float f) {
  unsigned i = __float_as_uint(f);
  return (i & 0x80000000u) ? ~i : (i | 0x80000000u);
}
__device__ __forceinline__ float ord2f(unsigned e) {
  unsigned i = (e & 0x80000000u) ? (e & 0x7FFFFFFFu) : ~e;
  return __uint_as_float(i);
}

// phase 1: a = leaky_relu(as[src]+ad[dst]); segment max over dst
__global__ void edge_p1(const int* __restrict__ src, const int* __restrict__ dst,
                        const float* __restrict__ as_, const float* __restrict__ ad_,
                        float* __restrict__ aE, unsigned* __restrict__ mEnc, int E) {
  int e = blockIdx.x * blockDim.x + threadIdx.x;
  if (e >= E) return;
  float a = as_[src[e]] + ad_[dst[e]];
  a = a > 0.f ? a : 0.2f * a;
  aE[e] = a;
  atomicMax(&mEnc[dst[e]], f2ord(a));
}
// phase 2: w = exp(a - m[dst]); segment sum z
__global__ void edge_p2(const int* __restrict__ dst, float* __restrict__ aE,
                        const unsigned* __restrict__ mEnc, float* __restrict__ z, int E) {
  int e = blockIdx.x * blockDim.x + threadIdx.x;
  if (e >= E) return;
  int d = dst[e];
  float w = __expf(aE[e] - ord2f(mEnc[d]));
  aE[e] = w;
  atomicAdd(&z[d], w);
}
// phase 3: acc[dst,:] += (w/z[dst]) * H[src,:]   (thread = (edge, channel))
__global__ void edge_p3(const int* __restrict__ src, const int* __restrict__ dst,
                        const float* __restrict__ aE, const float* __restrict__ z,
                        const float* __restrict__ H, float* __restrict__ acc,
                        int E, int C, int lc) {
  long i = blockIdx.x * (long)blockDim.x + threadIdx.x;
  if (i >= ((long)E << lc)) return;
  int e = (int)(i >> lc);
  int c = (int)(i & (C - 1));
  int s = src[e], d = dst[e];
  float alpha = aE[e] / z[d];
  atomicAdd(&acc[(long)d * C + c], alpha * H[(long)s * C + c]);
}

// x_new = relu((acc + sum(biases)) / relCount)
__global__ void finalize_x(const float* __restrict__ acc, const float* b0,
                           const float* b1, const float* b2, float invCnt,
                           float* __restrict__ xo, int N, int C) {
  long i = blockIdx.x * (long)blockDim.x + threadIdx.x;
  if (i >= (long)N * C) return;
  int c = (int)(i % C);
  float bs = b0[c];
  if (b1) bs += b1[c];
  if (b2) bs += b2[c];
  float v = (acc[i] + bs) * invCnt;
  xo[i] = v > 0.f ? v : 0.f;
}

// ---------------- pooling ----------------
__global__ void pool_count(const int* __restrict__ batch, float* __restrict__ cnt, int N) {
  int i = blockIdx.x * blockDim.x + threadIdx.x;
  if (i < N) atomicAdd(&cnt[batch[i]], 1.f);
}
__global__ void pool_sum(const float* __restrict__ x, const int* __restrict__ batch,
                         float* __restrict__ ps, int N, int C) {
  long i = blockIdx.x * (long)blockDim.x + threadIdx.x;
  if (i >= (long)N * C) return;
  int n = (int)(i / C), c = (int)(i % C);
  atomicAdd(&ps[(long)batch[n] * C + c], x[i]);
}
__global__ void pool_write(const float* __restrict__ ps, const float* __restrict__ cnt,
                           float* __restrict__ out, int C, int rowStride, int colOff) {
  int i = blockIdx.x * blockDim.x + threadIdx.x;
  if (i >= 64 * C) return;
  int g = i / C, c = i - g * C;
  float d = cnt[g] > 1.f ? cnt[g] : 1.f;
  out[(long)g * rowStride + colOff + c] = ps[i] / d;
}

// ============================= host orchestration =============================
extern "C" void kernel_launch(void* const* d_in, const int* in_sizes, int n_in,
                              void* d_out, int out_size, void* d_ws, size_t ws_size,
                              hipStream_t stream) {
  (void)in_sizes; (void)n_in; (void)out_size; (void)ws_size;
  const int NT[3]   = {100000, 100000, 10000};                  // node, edge, source
  const int EREL[5] = {800000, 800000, 200000, 800000, 200000}; // nn,en,sn,ne,ns
  const int LOUT[5] = {64, 128, 128, 128, 64};

  const float* x_in[3] = {(const float*)d_in[0], (const float*)d_in[1], (const float*)d_in[2]};
  const int* ei[5];
  for (int i = 0; i < 5; ++i) ei[i] = (const int*)d_in[3 + i];
  const int* batch[3] = {(const int*)d_in[8], (const int*)d_in[9], (const int*)d_in[10]};

  // params: 11 + layer*25 + relSortedIdx*5 + tensorIdx
  // relations sorted: en=0, ne=1, nn=2, ns=3, sn=4 ; tensors: W_dst,W_src,att_dst,att_src,b
  auto P = [&](int layer, int pIdx, int t) -> const float* {
    return (const float*)d_in[11 + layer * 25 + pIdx * 5 + t];
  };

  struct Rel { int s, d, eiIdx, pIdx; };
  const Rel rels[5] = {
    {0, 0, 0, 2},  // nn: node->node
    {1, 0, 1, 0},  // en: edge->node
    {2, 0, 2, 4},  // sn: source->node
    {0, 1, 3, 1},  // ne: node->edge
    {0, 2, 4, 3},  // ns: node->source
  };

  // ---- workspace carve-up (float units) ----
  float* w = (float*)d_ws;
  float* X[3]   = {w,             w + 12800000L, w + 25600000L};
  float* ACC[3] = {w + 26880000L, w + 39680000L, w + 52480000L};
  float* HSRC   = w + 53760000L;
  float* AS     = w + 66560000L;
  float* AD     = AS + 100000;
  float* AE     = AD + 100000;
  unsigned* MENC = (unsigned*)(AE + 800000);
  float* Z      = (float*)(MENC + 100000);
  float* WVEC   = Z + 100000;
  float* PS     = WVEC + 128;
  float* GCNT   = PS + 8192;
  bf16_t* XB[3] = {(bf16_t*)(w + 68000000L), (bf16_t*)(w + 74500000L),
                   (bf16_t*)(w + 81000000L)};
  bf16_t* WPACK = (bf16_t*)(w + 81700000L);

  const int B = 256;
  const float* xs[3] = {x_in[0], x_in[1], x_in[2]};
  int dIn = 32;
  long outOff = 0;

  for (int l = 0; l < 5; ++l) {
    int C  = LOUT[l];
    int lc = (C == 64) ? 6 : 7;

    // bf16 copies of current activations (amortized across all relations/tiles)
    for (int t = 0; t < 3; ++t)
      cvt_bf16<<<cdiv((long)NT[t] * dIn, B), B, 0, stream>>>(xs[t], XB[t], (long)NT[t] * dIn);

    for (int t = 0; t < 3; ++t)
      fill_f32<<<cdiv((long)NT[t] * C, B), B, 0, stream>>>(ACC[t], 0.f, NT[t] * C);

    int relCnt[3] = {0, 0, 0};
    for (int r = 0; r < 5; ++r) {
      const Rel& R = rels[r];
      int Ns = NT[R.s], Nd = NT[R.d], E = EREL[R.eiIdx];
      const int* esrc = ei[R.eiIdx];
      const int* edst = esrc + E;
      const float* Wsrc = P(l, R.pIdx, 1);
      const float* Wdst = P(l, R.pIdx, 0);
      const float* attD = P(l, R.pIdx, 2);
      const float* attS = P(l, R.pIdx, 3);

      // pack W_src into WMMA B layout (tiny), then h_src = x_src @ W_src via WMMA
      pack_w<<<cdiv((long)dIn * C, B), B, 0, stream>>>(Wsrc, WPACK, dIn, C);
      long waves = (long)(Ns / 16) * (C / 16);
      gemm_wmma<<<cdiv(waves * 32, B), B, 0, stream>>>(XB[R.s], WPACK, HSRC, Ns, dIn, C);
      // as = h_src . att_src
      dot_rows<<<cdiv((long)Ns * 32, B), B, 0, stream>>>(HSRC, attS, AS, Ns, C);
      // fold: wvec = W_dst @ att_dst ; ad = x_dst . wvec (skips full h_dst GEMM)
      dot_rows<<<cdiv((long)dIn * 32, B), B, 0, stream>>>(Wdst, attD, WVEC, dIn, C);
      dot_rows<<<cdiv((long)Nd * 32, B), B, 0, stream>>>(xs[R.d], WVEC, AD, Nd, dIn);

      fill_u32<<<cdiv(Nd, B), B, 0, stream>>>(MENC, 0u, Nd);
      fill_f32<<<cdiv(Nd, B), B, 0, stream>>>(Z, 0.f, Nd);
      edge_p1<<<cdiv(E, B), B, 0, stream>>>(esrc, edst, AS, AD, AE, MENC, E);
      edge_p2<<<cdiv(E, B), B, 0, stream>>>(edst, AE, MENC, Z, E);
      edge_p3<<<cdiv((long)E * C, B), B, 0, stream>>>(esrc, edst, AE, Z, HSRC, ACC[R.d], E, C, lc);
      relCnt[R.d]++;
    }

    int rowStride = 3 * C;
    for (int t = 0; t < 3; ++t) {
      const float* bs[3] = {nullptr, nullptr, nullptr};
      int nb = 0;
      for (int r = 0; r < 5; ++r)
        if (rels[r].d == t) bs[nb++] = P(l, rels[r].pIdx, 4);
      finalize_x<<<cdiv((long)NT[t] * C, B), B, 0, stream>>>(
          ACC[t], bs[0], bs[1], bs[2], 1.f / (float)relCnt[t], X[t], NT[t], C);

      fill_f32<<<cdiv(64 * C, B), B, 0, stream>>>(PS, 0.f, 64 * C);
      fill_f32<<<1, 64, 0, stream>>>(GCNT, 0.f, 64);
      pool_count<<<cdiv(NT[t], B), B, 0, stream>>>(batch[t], GCNT, NT[t]);
      pool_sum<<<cdiv((long)NT[t] * C, B), B, 0, stream>>>(X[t], batch[t], PS, NT[t], C);
      pool_write<<<cdiv(64 * C, B), B, 0, stream>>>(PS, GCNT, (float*)d_out + outOff, C,
                                                   rowStride, t * C);
    }

    xs[0] = X[0]; xs[1] = X[1]; xs[2] = X[2];
    dIn = C;
    outOff += 64L * rowStride;
  }
}
